// Model_12592844111975
// MI455X (gfx1250) — compile-verified, matrix operand order checked by reading, not run on hardware
//
#include <hip/hip_runtime.h>
#include <hip/hip_bf16.h>

typedef __attribute__((ext_vector_type(16))) _Float16 v16h;
typedef __attribute__((ext_vector_type(8)))  _Float16 v8h;
typedef __attribute__((ext_vector_type(8)))  float    v8f;

#define DEV __device__ __forceinline__

constexpr int   B_ = 128, L_ = 512, C_ = 256, H_ = 96;
constexpr int   P_ = 16, S_ = 8, N_ = 64, D_ = 32;
constexpr int   BC = B_ * C_;                 // 32768 rows
constexpr float ALPHA_ = 0.3f, EPS_ = 1e-5f;

DEV float gelu_f(float x)    { return 0.5f * x * (1.f + erff(x * 0.70710678118654752f)); }
DEV float sigmoid_f(float x) { return 1.f / (1.f + __expf(-x)); }

// ---------------------------------------------------------------------------
// Weight packing into WMMA B-fragment tiles (f32 -> f16).
// Tile (kt, nt): 32 lanes x 16 halves; lane's k pattern matches the A-side:
//   k = kt*32 + (lane>=16 ? 8 : 0) + (i&7) + (i>>3)*16,  n = nt*16 + (lane&15)
// mode 0: B[k][n] = W[k*stride + n]
// mode 1: mW1 row-permute: B[k][n] = W[((k&63)*32 + (k>>6))*stride + n]
// mode 2: transpose (pwW^T): B[k][n] = W[n*stride + k]
// ---------------------------------------------------------------------------
__global__ __launch_bounds__(256) void pack_b_kernel(
    const float* __restrict__ W, _Float16* __restrict__ out,
    int Ksrc, int KT, int NT, int stride, int mode)
{
    size_t idx = (size_t)blockIdx.x * 256 + threadIdx.x;
    size_t total = (size_t)KT * NT * 512;
    if (idx >= total) return;
    int    i    = (int)(idx & 15);
    int    lane = (int)((idx >> 4) & 31);
    size_t tile = idx >> 9;
    int    nt   = (int)(tile % NT);
    int    kt   = (int)(tile / NT);
    int k = kt * 32 + ((lane >> 4) << 3) + (i & 7) + ((i >> 3) << 4);
    int n = nt * 16 + (lane & 15);
    float v = 0.f;
    if (k < Ksrc) {
        if (mode == 0)      v = W[(size_t)k * stride + n];
        else if (mode == 1) { int ks = (k & 63) * 32 + (k >> 6); v = W[(size_t)ks * stride + n]; }
        else                v = W[(size_t)n * stride + k];
    }
    out[idx] = (_Float16)v;
}

// ---------------------------------------------------------------------------
// Generic WMMA GEMM: C[M x N] = A[M x K](f16,row-major) * Bpacked + bias.
// Block = 8 waves; wave computes a 16 x (CT*16) strip. grid = (M/128, NT/CT).
// K-loop unrolled x2 with ping-pong fragment sets: each set's registers are
// refilled by loads in place right after its WMMAs consume it, so loads stay
// one step ahead of the matrix ops with zero register-rotation copies.
// Requires KT (= K/32) even, or KT == 1 (single-shot path).
// EPI: 0 = store f32, 1 = gelu->f16, 2 = store f16, 4 = sigmoid->f32
// ---------------------------------------------------------------------------
template<int CT, int EPI>
__global__ __launch_bounds__(256) void gemm_wmma_kernel(
    const _Float16* __restrict__ A, int lda,
    const _Float16* __restrict__ Bp, int K, int NT,
    const float* __restrict__ bias,
    float* __restrict__ Cf, _Float16* __restrict__ Ch, int ldc, int col0)
{
    const int    lane = threadIdx.x & 31;
    const int    wave = threadIdx.x >> 5;
    const size_t r0   = (size_t)blockIdx.x * 128 + (size_t)wave * 16;
    const int    nt0  = blockIdx.y * CT;

    v8f acc[CT];
#pragma unroll
    for (int c = 0; c < CT; ++c)
#pragma unroll
        for (int t = 0; t < 8; ++t) acc[c][t] = 0.f;

    const size_t abase = (r0 + (size_t)(lane & 15)) * (size_t)lda + (size_t)((lane >> 4) << 3);
    const _Float16* bbase = Bp + ((size_t)(nt0)*32 + (size_t)lane) * 16;
    const int KT = K >> 5;

    auto load_a = [&](int kt) -> v16h {
        const _Float16* ap = A + abase + (size_t)kt * 32;
        v8h lo = *reinterpret_cast<const v8h*>(ap);
        v8h hi = *reinterpret_cast<const v8h*>(ap + 16);
        v16h a;
#pragma unroll
        for (int i = 0; i < 8; ++i) { a[i] = lo[i]; a[i + 8] = hi[i]; }
        return a;
    };
    auto load_b = [&](int kt, int c) -> v16h {
        return *reinterpret_cast<const v16h*>(bbase + ((size_t)kt * NT + c) * 512);
    };
    auto do_wmma = [&](const v16h& a, const v16h (&b)[CT]) {
#pragma unroll
        for (int c = 0; c < CT; ++c)
            acc[c] = __builtin_amdgcn_wmma_f32_16x16x32_f16(
                false, a, false, b[c], (short)0, acc[c], false, false);
    };

    v16h a0, a1;
    v16h b0[CT], b1[CT];

    a0 = load_a(0);
#pragma unroll
    for (int c = 0; c < CT; ++c) b0[c] = load_b(0, c);

    if (KT == 1) {
        do_wmma(a0, b0);
    } else {
        a1 = load_a(1);
#pragma unroll
        for (int c = 0; c < CT; ++c) b1[c] = load_b(1, c);

        for (int kt = 0; kt + 2 < KT; kt += 2) {
            do_wmma(a0, b0);                 // consume set 0 (kt)
            a0 = load_a(kt + 2);             // refill set 0 in place (kt+2)
#pragma unroll
            for (int c = 0; c < CT; ++c) b0[c] = load_b(kt + 2, c);
            do_wmma(a1, b1);                 // consume set 1 (kt+1)
            a1 = load_a(kt + 3);             // refill set 1 in place (kt+3)
#pragma unroll
            for (int c = 0; c < CT; ++c) b1[c] = load_b(kt + 3, c);
        }
        do_wmma(a0, b0);                     // kt = KT-2
        do_wmma(a1, b1);                     // kt = KT-1
    }

    const int col_lane = lane & 15;
    const int rowoff   = (lane >> 4) * 8;
#pragma unroll
    for (int c = 0; c < CT; ++c) {
        const int   col = (nt0 + c) * 16 + col_lane;
        const float bv  = bias ? bias[col] : 0.f;
#pragma unroll
        for (int v = 0; v < 8; ++v) {
            const size_t r   = r0 + (size_t)(rowoff + v);
            float        val = acc[c][v] + bv;
            const size_t o   = r * (size_t)ldc + (size_t)(col0 + col);
            if (EPI == 0)      Cf[o] = val;
            else if (EPI == 1) Ch[o] = (_Float16)gelu_f(val);
            else if (EPI == 2) Ch[o] = (_Float16)val;
            else if (EPI == 4) Cf[o] = sigmoid_f(val);
        }
    }
}

// ------------------------- RevIN statistics (ddof=1) ------------------------
__global__ __launch_bounds__(256) void revin_stats_kernel(
    const float* __restrict__ x, float* __restrict__ meanb, float* __restrict__ stdb)
{
    int b = blockIdx.x, c = threadIdx.x;
    const float* px = x + (size_t)b * L_ * C_ + c;
    float s = 0.f, sq = 0.f;
    for (int l = 0; l < L_; ++l) { float v = px[(size_t)l * C_]; s += v; sq += v * v; }
    float m   = s / (float)L_;
    float var = (sq - (float)L_ * m * m) / (float)(L_ - 1);
    meanb[b * C_ + c] = m;
    stdb[b * C_ + c]  = sqrtf(fmaxf(var, 0.f)) + EPS_;
}

// --------------- normalize + sequential EMA trend / seasonal ---------------
__global__ __launch_bounds__(256) void norm_ema_kernel(
    const float* __restrict__ x, const float* __restrict__ meanb,
    const float* __restrict__ stdb, const float* __restrict__ rw,
    const float* __restrict__ rb, _Float16* __restrict__ t_out,
    float* __restrict__ s_out)
{
    int b = blockIdx.x, c = threadIdx.x;
    const size_t bc = (size_t)b * C_ + c;
    const float m = meanb[bc], sd = stdb[bc], w = rw[c], bb = rb[c];
    const float* px = x + (size_t)b * L_ * C_ + c;
    float trend = 0.f;
    for (int l = 0; l < L_; ++l) {
        float xn = (px[(size_t)l * C_] - m) / sd * w + bb;
        trend = (l == 0) ? xn : (ALPHA_ * xn + (1.f - ALPHA_) * trend);
        t_out[bc * L_ + l] = (_Float16)trend;
        s_out[bc * L_ + l] = xn - trend;
    }
}

// ---- seasonal 3-tap conv + edge-padded patch extraction (K padded to 32) ---
__global__ __launch_bounds__(256) void season_patch_kernel(
    const float* __restrict__ s_in, const float* __restrict__ aggw,
    _Float16* __restrict__ patches)
{
    __shared__ float sh[L_];
    __shared__ float sh2[L_];
    const size_t bc = blockIdx.x;
    const int    t  = threadIdx.x;
    const float w0 = aggw[0], w1 = aggw[1], w2 = aggw[2];
    for (int i = t; i < L_; i += 256) sh[i] = s_in[bc * L_ + i];
    __syncthreads();
    for (int i = t; i < L_; i += 256) {
        float a  = (i > 0)      ? sh[i - 1] : 0.f;
        float bm = sh[i];
        float cc = (i < L_ - 1) ? sh[i + 1] : 0.f;
        sh2[i] = bm + (w0 * a + w1 * bm + w2 * cc);
    }
    __syncthreads();
    _Float16* pp = patches + bc * (size_t)(N_ * 32);
    for (int j = t; j < N_ * 32; j += 256) {
        int n = j >> 5, p = j & 31;
        float v = 0.f;
        if (p < P_) { int idx = n * S_ + p; if (idx > L_ - 1) idx = L_ - 1; v = sh2[idx]; }
        pp[j] = (_Float16)v;
    }
}

// ----------------------------- LayerNorm rows -------------------------------
__global__ __launch_bounds__(256) void ln_rows_kernel(
    const float* __restrict__ x, int W, const float* __restrict__ g,
    const float* __restrict__ b, _Float16* __restrict__ out)
{
    const int    lane = threadIdx.x & 31;
    const int    wave = threadIdx.x >> 5;
    const size_t row  = (size_t)blockIdx.x * 8 + wave;
    const int    per  = W >> 5;            // <= 8
    const float* xr = x + row * (size_t)W;
    float v[8];
    float s = 0.f, sq = 0.f;
    for (int i = 0; i < per; ++i) { v[i] = xr[lane + i * 32]; s += v[i]; sq += v[i] * v[i]; }
    for (int o = 16; o > 0; o >>= 1) { s += __shfl_xor(s, o, 32); sq += __shfl_xor(sq, o, 32); }
    const float mean = s / (float)W;
    const float inv  = rsqrtf(sq / (float)W - mean * mean + EPS_);
    _Float16* orow = out + row * (size_t)W;
    for (int i = 0; i < per; ++i) {
        int j = lane + i * 32;
        orow[j] = (_Float16)((v[i] - mean) * inv * g[j] + b[j]);
    }
}

// ---------------- BatchNorm stats (global reduction via atomics) ------------
// mode 0: layout (bc, n, d) -> channel = (i>>5)&63 ; mode 1: (bc, d, o) -> i&63
__global__ __launch_bounds__(256) void bn_stats_kernel(
    const _Float16* __restrict__ x, size_t total, int mode, float* __restrict__ acc)
{
    __shared__ float ssum[64], ssq[64];
    if (threadIdx.x < 64) { ssum[threadIdx.x] = 0.f; ssq[threadIdx.x] = 0.f; }
    __syncthreads();
    const size_t stride = (size_t)gridDim.x * blockDim.x;
    for (size_t i = (size_t)blockIdx.x * blockDim.x + threadIdx.x; i < total; i += stride) {
        float v  = (float)x[i];
        int   ch = mode ? (int)(i & 63) : (int)((i >> 5) & 63);
        atomicAdd(&ssum[ch], v);
        atomicAdd(&ssq[ch], v * v);
    }
    __syncthreads();
    if (threadIdx.x < 64) {
        atomicAdd(&acc[threadIdx.x],      ssum[threadIdx.x]);
        atomicAdd(&acc[64 + threadIdx.x], ssq[threadIdx.x]);
    }
}

__global__ void bn_finalize_kernel(const float* __restrict__ acc,
    const float* __restrict__ g, const float* __restrict__ b,
    float* __restrict__ ab, float cnt)
{
    int ch = threadIdx.x;
    if (ch < 64) {
        float m   = acc[ch] / cnt;
        float var = acc[64 + ch] / cnt - m * m;
        float inv = rsqrtf(var + EPS_);
        float A   = g[ch] * inv;
        ab[ch]      = A;
        ab[64 + ch] = b[ch] - m * A;
    }
}

__global__ __launch_bounds__(256) void bn_gelu_kernel(
    const _Float16* __restrict__ x, const float* __restrict__ ab,
    const _Float16* __restrict__ res, _Float16* __restrict__ out,
    size_t total, int mode)
{
    const size_t stride = (size_t)gridDim.x * blockDim.x;
    for (size_t i = (size_t)blockIdx.x * blockDim.x + threadIdx.x; i < total; i += stride) {
        int   ch = mode ? (int)(i & 63) : (int)((i >> 5) & 63);
        float v  = gelu_f(ab[ch] * (float)x[i] + ab[64 + ch]);
        if (res) v += (float)res[i];
        out[i] = (_Float16)v;
    }
}

// ------------------ multi-scale depthwise conv along D ---------------------
__global__ __launch_bounds__(256) void dwconv_kernel(
    const _Float16* __restrict__ e,
    const float* __restrict__ dw1, const float* __restrict__ dw3,
    const float* __restrict__ dw5, const float* __restrict__ dw7,
    _Float16* __restrict__ out, size_t total)
{
    const size_t stride = (size_t)gridDim.x * blockDim.x;
    for (size_t i = (size_t)blockIdx.x * blockDim.x + threadIdx.x; i < total; i += stride) {
        int d = (int)(i & 31);
        int n = (int)((i >> 5) & 63);
        const _Float16* row = e + (i - (size_t)d);
        float acc = dw1[n] * (float)row[d];
#pragma unroll
        for (int j = 0; j < 3; ++j) { int dd = d + j - 1; if (dd >= 0 && dd < 32) acc += dw3[n * 3 + j] * (float)row[dd]; }
#pragma unroll
        for (int j = 0; j < 5; ++j) { int dd = d + j - 2; if (dd >= 0 && dd < 32) acc += dw5[n * 5 + j] * (float)row[dd]; }
#pragma unroll
        for (int j = 0; j < 7; ++j) { int dd = d + j - 3; if (dd >= 0 && dd < 32) acc += dw7[n * 7 + j] * (float)row[dd]; }
        out[i] = (_Float16)acc;
    }
}

// ------------------------- mean over D=32 (gate input) ----------------------
__global__ __launch_bounds__(256) void mean_d_kernel(
    const _Float16* __restrict__ x, _Float16* __restrict__ out, size_t rows)
{
    const size_t stride = (size_t)gridDim.x * blockDim.x;
    for (size_t r = (size_t)blockIdx.x * blockDim.x + threadIdx.x; r < rows; r += stride) {
        const _Float16* p = x + r * 32;
        float s = 0.f;
#pragma unroll
        for (int d = 0; d < 32; ++d) s += (float)p[d];
        out[r] = (_Float16)(s * (1.f / 32.f));
    }
}

// ------------- gate multiply + transpose (bc,n,d) -> (bc,d,n) ---------------
__global__ __launch_bounds__(256) void gate_mul_t_kernel(
    const _Float16* __restrict__ out1, const float* __restrict__ gate,
    _Float16* __restrict__ gatedT, size_t total)
{
    const size_t stride = (size_t)gridDim.x * blockDim.x;
    for (size_t i = (size_t)blockIdx.x * blockDim.x + threadIdx.x; i < total; i += stride) {
        size_t bc = i >> 11;
        int    n  = (int)((i >> 5) & 63);
        int    d  = (int)(i & 31);
        gatedT[(bc << 11) + (size_t)d * 64 + n] =
            (_Float16)((float)out1[i] * gate[bc * 64 + n]);
    }
}

// ------------------------------ final denorm --------------------------------
__global__ __launch_bounds__(256) void denorm_kernel(
    const float* __restrict__ y, const float* __restrict__ rw,
    const float* __restrict__ rb, const float* __restrict__ meanb,
    const float* __restrict__ stdb, float* __restrict__ out, size_t total)
{
    const size_t stride = (size_t)gridDim.x * blockDim.x;
    for (size_t i = (size_t)blockIdx.x * blockDim.x + threadIdx.x; i < total; i += stride) {
        int    c  = (int)(i % C_);
        size_t t2 = i / C_;
        int    h  = (int)(t2 % H_);
        int    b  = (int)(t2 / H_);
        size_t bc = (size_t)b * C_ + c;
        out[i] = (y[bc * H_ + h] - rb[c]) / (rw[c] + EPS_) * stdb[bc] + meanb[bc];
    }
}

__global__ void fill_zero_kernel(float* __restrict__ p, int n)
{
    int i = blockIdx.x * blockDim.x + threadIdx.x;
    if (i < n) p[i] = 0.f;
}

// ===========================================================================
extern "C" void kernel_launch(void* const* d_in, const int* in_sizes, int n_in,
                              void* d_out, int out_size, void* d_ws, size_t ws_size,
                              hipStream_t stream)
{
    (void)in_sizes; (void)n_in; (void)out_size; (void)ws_size;

    const float* x       = (const float*)d_in[0];
    const float* revin_w = (const float*)d_in[1];
    const float* revin_b = (const float*)d_in[2];
    const float* lsW1    = (const float*)d_in[3];
    const float* lsb1    = (const float*)d_in[4];
    const float* ln1w    = (const float*)d_in[5];
    const float* ln1b    = (const float*)d_in[6];
    const float* lsW2    = (const float*)d_in[7];
    const float* lsb2    = (const float*)d_in[8];
    const float* ln2w    = (const float*)d_in[9];
    const float* ln2b    = (const float*)d_in[10];
    const float* lsW3    = (const float*)d_in[11];
    const float* lsb3    = (const float*)d_in[12];
    const float* aggw    = (const float*)d_in[13];
    const float* embW    = (const float*)d_in[14];
    const float* embb    = (const float*)d_in[15];
    const float* ebng    = (const float*)d_in[16];
    const float* ebnb    = (const float*)d_in[17];
    const float* dw1     = (const float*)d_in[18];
    const float* dw3     = (const float*)d_in[19];
    const float* dw5     = (const float*)d_in[20];
    const float* dw7     = (const float*)d_in[21];
    const float* msg     = (const float*)d_in[22];
    const float* msb     = (const float*)d_in[23];
    const float* gW1     = (const float*)d_in[24];
    const float* gb1     = (const float*)d_in[25];
    const float* gW2     = (const float*)d_in[26];
    const float* gb2     = (const float*)d_in[27];
    const float* pwW     = (const float*)d_in[28];
    const float* pwb     = (const float*)d_in[29];
    const float* pwg     = (const float*)d_in[30];
    const float* pwbnb   = (const float*)d_in[31];
    const float* mW1     = (const float*)d_in[32];
    const float* mb1     = (const float*)d_in[33];
    const float* mW2     = (const float*)d_in[34];
    const float* mb2     = (const float*)d_in[35];
    const float* fW      = (const float*)d_in[36];
    const float* fb      = (const float*)d_in[37];

    // ---- workspace carve-out (256B aligned regions) ----
    char*  base = (char*)d_ws;
    size_t off  = 0;
    auto alloc = [&](size_t bytes) -> char* {
        char* p = base + off;
        off = (off + bytes + 255) & ~(size_t)255;
        return p;
    };
    const size_t ETOT = (size_t)BC * 2048;   // (bc, 64, 32) element count

    float*    meanb   = (float*)   alloc((size_t)BC * 4);
    float*    stdb    = (float*)   alloc((size_t)BC * 4);
    _Float16* t16     = (_Float16*)alloc((size_t)BC * L_ * 2);
    float*    s32     = (float*)   alloc((size_t)BC * L_ * 4);
    _Float16* patches = (_Float16*)alloc(ETOT * 2);
    float*    h1f     = (float*)   alloc((size_t)BC * 256 * 4);
    _Float16* h1h     = (_Float16*)alloc((size_t)BC * 256 * 2);
    float*    h2f     = (float*)   alloc((size_t)BC * 128 * 4);
    _Float16* h2h     = (_Float16*)alloc((size_t)BC * 128 * 2);
    _Float16* cat     = (_Float16*)alloc((size_t)BC * 192 * 2);
    _Float16* eraw    = (_Float16*)alloc(ETOT * 2);
    _Float16* e       = (_Float16*)alloc(ETOT * 2);
    _Float16* convraw = (_Float16*)alloc(ETOT * 2);
    _Float16* out1    = (_Float16*)alloc(ETOT * 2);
    _Float16* gatedT  = (_Float16*)alloc(ETOT * 2);
    _Float16* Gin     = (_Float16*)alloc((size_t)BC * 64 * 2);
    _Float16* g1      = (_Float16*)alloc((size_t)BC * 128 * 2);
    float*    gate    = (float*)   alloc((size_t)BC * 64 * 4);
    _Float16* m1      = (_Float16*)alloc((size_t)BC * 192 * 2);
    float*    yv      = (float*)   alloc((size_t)BC * 96 * 4);
    float*    bnacc   = (float*)   alloc(128 * 4);
    float*    bnab    = (float*)   alloc(128 * 4);
    _Float16* lsW1p   = (_Float16*)alloc((size_t)16 * 16 * 512 * 2);
    _Float16* lsW2p   = (_Float16*)alloc((size_t)8  * 8  * 512 * 2);
    _Float16* lsW3p   = (_Float16*)alloc((size_t)4  * 6  * 512 * 2);
    _Float16* embWp   = (_Float16*)alloc((size_t)1  * 2  * 512 * 2);
    _Float16* gW1p    = (_Float16*)alloc((size_t)2  * 8  * 512 * 2);
    _Float16* gW2p    = (_Float16*)alloc((size_t)4  * 4  * 512 * 2);
    _Float16* pwWTp   = (_Float16*)alloc((size_t)2  * 4  * 512 * 2);
    _Float16* mW1p    = (_Float16*)alloc((size_t)64 * 12 * 512 * 2);
    _Float16* mW2p    = (_Float16*)alloc((size_t)6  * 6  * 512 * 2);
    _Float16* fWp     = (_Float16*)alloc((size_t)6  * 6  * 512 * 2);
    _Float16* out2    = eraw;     // reuse after e-chain no longer needs raw
    _Float16* out3    = convraw;  // reuse

    dim3 blk(256);
    auto pack = [&](const float* W, _Float16* o, int Ksrc, int KT, int NT, int st, int mode) {
        size_t total = (size_t)KT * NT * 512;
        pack_b_kernel<<<dim3((unsigned)((total + 255) / 256)), blk, 0, stream>>>(W, o, Ksrc, KT, NT, st, mode);
    };
    pack(lsW1, lsW1p,  512, 16, 16, 256, 0);
    pack(lsW2, lsW2p,  256,  8,  8, 128, 0);
    pack(lsW3, lsW3p,  128,  4,  6,  96, 0);
    pack(embW, embWp,   16,  1,  2,  32, 0);
    pack(gW1,  gW1p,    64,  2,  8, 128, 0);
    pack(gW2,  gW2p,   128,  4,  4,  64, 0);
    pack(pwW,  pwWTp,   64,  2,  4,  64, 2);   // transpose: B = pwW^T
    pack(mW1,  mW1p,  2048, 64, 12, 192, 1);   // row-permuted for (bc,d,o) layout
    pack(mW2,  mW2p,   192,  6,  6,  96, 0);
    pack(fW,   fWp,    192,  6,  6,  96, 0);

    // RevIN stats -> normalize + EMA -> seasonal conv + patches
    revin_stats_kernel<<<dim3(B_), blk, 0, stream>>>(x, meanb, stdb);
    norm_ema_kernel<<<dim3(B_), blk, 0, stream>>>(x, meanb, stdb, revin_w, revin_b, t16, s32);
    season_patch_kernel<<<dim3(BC), blk, 0, stream>>>(s32, aggw, patches);

    // Trend MLP: (t @ lsW1 -> LN) -> (@ lsW2 -> LN) -> (@ lsW3 -> cat[:, :96])
    gemm_wmma_kernel<4, 0><<<dim3(BC / 128, 4), blk, 0, stream>>>(t16, 512, lsW1p, 512, 16, lsb1, h1f, nullptr, 256, 0);
    ln_rows_kernel<<<dim3(BC / 8), blk, 0, stream>>>(h1f, 256, ln1w, ln1b, h1h);
    gemm_wmma_kernel<4, 0><<<dim3(BC / 128, 2), blk, 0, stream>>>(h1h, 256, lsW2p, 256, 8, lsb2, h2f, nullptr, 128, 0);
    ln_rows_kernel<<<dim3(BC / 8), blk, 0, stream>>>(h2f, 128, ln2w, ln2b, h2h);
    gemm_wmma_kernel<2, 2><<<dim3(BC / 128, 3), blk, 0, stream>>>(h2h, 128, lsW3p, 128, 6, lsb3, nullptr, cat, 192, 0);

    // Patch embedding GEMM (M = BC*64, K padded 16->32, single K-tile path)
    gemm_wmma_kernel<2, 2><<<dim3((BC * 64) / 128, 1), blk, 0, stream>>>(patches, 32, embWp, 32, 2, embb, nullptr, eraw, 32, 0);

    // BN1 + gelu -> e
    fill_zero_kernel<<<dim3(1), dim3(128), 0, stream>>>(bnacc, 128);
    bn_stats_kernel<<<dim3(2048), blk, 0, stream>>>(eraw, ETOT, 0, bnacc);
    bn_finalize_kernel<<<dim3(1), dim3(64), 0, stream>>>(bnacc, ebng, ebnb, bnab, (float)((size_t)BC * 32));
    bn_gelu_kernel<<<dim3(32768), blk, 0, stream>>>(eraw, bnab, nullptr, e, ETOT, 0);

    // Multi-scale depthwise conv -> BN2 + gelu + residual -> out1
    dwconv_kernel<<<dim3(32768), blk, 0, stream>>>(e, dw1, dw3, dw5, dw7, convraw, ETOT);
    fill_zero_kernel<<<dim3(1), dim3(128), 0, stream>>>(bnacc, 128);
    bn_stats_kernel<<<dim3(2048), blk, 0, stream>>>(convraw, ETOT, 0, bnacc);
    bn_finalize_kernel<<<dim3(1), dim3(64), 0, stream>>>(bnacc, msg, msb, bnab, (float)((size_t)BC * 32));
    bn_gelu_kernel<<<dim3(32768), blk, 0, stream>>>(convraw, bnab, e, out1, ETOT, 0);

    // Gate MLP: mean_d -> gW1(gelu) -> gW2(sigmoid)
    mean_d_kernel<<<dim3(8192), blk, 0, stream>>>(out1, Gin, (size_t)BC * 64);
    gemm_wmma_kernel<4, 1><<<dim3(BC / 128, 2), blk, 0, stream>>>(Gin, 64, gW1p, 64, 8, gb1, nullptr, g1, 128, 0);
    gemm_wmma_kernel<4, 4><<<dim3(BC / 128, 1), blk, 0, stream>>>(g1, 128, gW2p, 128, 4, gb2, gate, nullptr, 64, 0);

    // gated^T (bc,d,n) then pointwise channel mix as one GEMM (M = BC*32)
    gate_mul_t_kernel<<<dim3(32768), blk, 0, stream>>>(out1, gate, gatedT, ETOT);
    gemm_wmma_kernel<4, 2><<<dim3((BC * 32) / 128, 1), blk, 0, stream>>>(gatedT, 64, pwWTp, 64, 4, pwb, nullptr, out2, 64, 0);

    // BN3 + gelu -> out3 (layout (bc,d,o); mW1 pre-permuted to match)
    fill_zero_kernel<<<dim3(1), dim3(128), 0, stream>>>(bnacc, 128);
    bn_stats_kernel<<<dim3(2048), blk, 0, stream>>>(out2, ETOT, 1, bnacc);
    bn_finalize_kernel<<<dim3(1), dim3(64), 0, stream>>>(bnacc, pwg, pwbnb, bnab, (float)((size_t)BC * 32));
    bn_gelu_kernel<<<dim3(32768), blk, 0, stream>>>(out2, bnab, nullptr, out3, ETOT, 1);

    // Seasonal head: mW1(gelu) -> mW2 -> cat[:, 96:192]
    gemm_wmma_kernel<4, 1><<<dim3(BC / 128, 3), blk, 0, stream>>>(out3, 2048, mW1p, 2048, 12, mb1, nullptr, m1, 192, 0);
    gemm_wmma_kernel<2, 2><<<dim3(BC / 128, 3), blk, 0, stream>>>(m1, 192, mW2p, 192, 6, mb2, nullptr, cat, 192, 96);

    // Fusion head + denorm
    gemm_wmma_kernel<2, 0><<<dim3(BC / 128, 3), blk, 0, stream>>>(cat, 192, fWp, 192, 6, fb, yv, nullptr, 96, 0);
    denorm_kernel<<<dim3(12288), blk, 0, stream>>>(yv, revin_w, revin_b, meanb, stdb,
                                                   (float*)d_out, (size_t)B_ * H_ * C_);
}
